// QuantizedLinear_56882546868863
// MI455X (gfx1250) — compile-verified, matrix operand order checked by reading, not run on hardware
//
#include <hip/hip_runtime.h>

typedef __attribute__((ext_vector_type(8))) int v8i;
typedef int v4i_vs __attribute__((vector_size(16)));

#define M_DIM 8192   // B*S = 4*2048
#define K_DIM 4096   // Din
#define N_DIM 4096   // Dout
#define NPART 1024
#define NIT   (K_DIM / 64)

#if __has_builtin(__builtin_amdgcn_global_load_async_to_lds_b128)
#define HAVE_ASYNC_LDS 1
#endif

__device__ __forceinline__ void async_copy16(const void* g, void* l) {
#ifdef HAVE_ASYNC_LDS
    // global -> LDS DMA, tracked by ASYNCcnt, no staging VGPRs
    __attribute__((address_space(1))) v4i_vs* gp =
        (__attribute__((address_space(1))) v4i_vs*)const_cast<void*>(g);
    __attribute__((address_space(3))) v4i_vs* lp =
        (__attribute__((address_space(3))) v4i_vs*)l;
    __builtin_amdgcn_global_load_async_to_lds_b128(gp, lp, 0, 0);
#else
    *(uint4*)l = *(const uint4*)g;
#endif
}

template <int N>
__device__ __forceinline__ void wait_async() {
#ifdef HAVE_ASYNC_LDS
#if __has_builtin(__builtin_amdgcn_s_wait_asynccnt)
    __builtin_amdgcn_s_wait_asynccnt(N);
#else
    asm volatile("s_wait_asynccnt %0" ::"i"(N) : "memory");
#endif
#endif
}

__device__ __forceinline__ int clamp_i8(float v) {
    int i = (int)rintf(v);
    i = i < -128 ? -128 : i;
    i = i > 127 ? 127 : i;
    return i;
}

// ---------------- Pass 1: per-block min/max partials ----------------
__global__ __launch_bounds__(256) void k_minmax(const float* __restrict__ x, int n4,
                                                float* __restrict__ partials) {
    __shared__ float smin[256], smax[256];
    float mn = 3.4e38f, mx = -3.4e38f;
    const float4* x4 = (const float4*)x;
    for (int i = blockIdx.x * blockDim.x + threadIdx.x; i < n4; i += gridDim.x * blockDim.x) {
        float4 v = x4[i];
        mn = fminf(mn, fminf(fminf(v.x, v.y), fminf(v.z, v.w)));
        mx = fmaxf(mx, fmaxf(fmaxf(v.x, v.y), fmaxf(v.z, v.w)));
    }
    smin[threadIdx.x] = mn; smax[threadIdx.x] = mx;
    __syncthreads();
    for (int s = 128; s > 0; s >>= 1) {
        if (threadIdx.x < s) {
            smin[threadIdx.x] = fminf(smin[threadIdx.x], smin[threadIdx.x + s]);
            smax[threadIdx.x] = fmaxf(smax[threadIdx.x], smax[threadIdx.x + s]);
        }
        __syncthreads();
    }
    if (threadIdx.x == 0) {
        partials[2 * blockIdx.x]     = smin[0];
        partials[2 * blockIdx.x + 1] = smax[0];
    }
}

// ---------------- Pass 2: finalize scale / zero-point ----------------
__global__ __launch_bounds__(256) void k_finalize(const float* __restrict__ partials, int np,
                                                  float* __restrict__ scalezp) {
    __shared__ float smin[256], smax[256];
    float mn = 3.4e38f, mx = -3.4e38f;
    for (int i = threadIdx.x; i < np; i += 256) {
        mn = fminf(mn, partials[2 * i]);
        mx = fmaxf(mx, partials[2 * i + 1]);
    }
    smin[threadIdx.x] = mn; smax[threadIdx.x] = mx;
    __syncthreads();
    for (int s = 128; s > 0; s >>= 1) {
        if (threadIdx.x < s) {
            smin[threadIdx.x] = fminf(smin[threadIdx.x], smin[threadIdx.x + s]);
            smax[threadIdx.x] = fmaxf(smax[threadIdx.x], smax[threadIdx.x + s]);
        }
        __syncthreads();
    }
    if (threadIdx.x == 0) {
        float fmn = smin[0], fmx = smax[0];
        float scale = (fmx - fmn) / 255.0f;
        float zp = rintf(-128.0f - fmn / scale);
        zp = fminf(127.0f, fmaxf(-128.0f, zp));
        scalezp[0] = scale;
        scalezp[1] = zp;
    }
}

// ---------------- Pass 3: quantize activations to packed int8 ----------------
__global__ __launch_bounds__(256) void k_quant_x(const float* __restrict__ x,
                                                 const float* __restrict__ scalezp,
                                                 int n4, int* __restrict__ q8) {
    const float inv = 1.0f / scalezp[0];
    const float zp  = scalezp[1];
    const float4* x4 = (const float4*)x;
    for (int i = blockIdx.x * blockDim.x + threadIdx.x; i < n4; i += gridDim.x * blockDim.x) {
        float4 v = x4[i];
        int a = clamp_i8(fmaf(v.x, inv, zp));
        int b = clamp_i8(fmaf(v.y, inv, zp));
        int c = clamp_i8(fmaf(v.z, inv, zp));
        int d = clamp_i8(fmaf(v.w, inv, zp));
        q8[i] = (a & 255) | ((b & 255) << 8) | ((c & 255) << 16) | ((d & 255) << 24);
    }
}

// ---------------- Pass 4: quantize weights to int8 + per-row sums ----------------
__global__ __launch_bounds__(256) void k_quant_w(const float* __restrict__ w,
                                                 int* __restrict__ w8,
                                                 int* __restrict__ rowsum) {
    __shared__ int ssum[256];
    const int row = blockIdx.x;
    const int tid = threadIdx.x;
    const float* wrow = w + (size_t)row * K_DIM;
    int* w8row = w8 + (size_t)row * (K_DIM / 4);
    int sum = 0;
    int packed[4];
    #pragma unroll
    for (int j = 0; j < 4; ++j) {
        float4 v = *(const float4*)(wrow + tid * 16 + j * 4);
        int a = (int)rintf(v.x), b = (int)rintf(v.y), c = (int)rintf(v.z), d = (int)rintf(v.w);
        sum += a + b + c + d;
        packed[j] = (a & 255) | ((b & 255) << 8) | ((c & 255) << 16) | ((d & 255) << 24);
    }
    *(int4*)(w8row + tid * 4) = make_int4(packed[0], packed[1], packed[2], packed[3]);
    ssum[tid] = sum;
    __syncthreads();
    for (int s = 128; s > 0; s >>= 1) {
        if (tid < s) ssum[tid] += ssum[tid + s];
        __syncthreads();
    }
    if (tid == 0) rowsum[row] = ssum[0];
}

// ---------------- Pass 5: int8 WMMA GEMM + dequant epilogue ----------------
// Block tile: 128(M) x 256(N), K-step 64. 8 waves: 2(M) x 4(N); wave tile 64x64.
// Triple-buffered LDS, 2-deep async global->LDS prefetch with partial
// s_wait_asynccnt (<=6 keeps the newest tile copy in flight).
__global__ __launch_bounds__(256) void k_gemm(const char* __restrict__ q8,
                                              const char* __restrict__ w8,
                                              const int* __restrict__ rowsum,
                                              const float* __restrict__ scalezp,
                                              const float* __restrict__ wscale,
                                              const float* __restrict__ bias,
                                              float* __restrict__ out) {
    __shared__ __align__(16) char As[3][128 * 64];   // 3 x 8 KB
    __shared__ __align__(16) char Bs[3][256 * 64];   // 3 x 16 KB

    const int tid   = threadIdx.x;
    const int lane  = tid & 31;
    const int wv    = tid >> 5;
    const int waveM = wv & 1;        // 0..1
    const int waveN = wv >> 1;       // 0..3
    const int half  = lane >> 4;     // 0/1
    const int l16   = lane & 15;

    const int mBase = blockIdx.y * 128;
    const int nBase = blockIdx.x * 256;

    const v8i vzero = {0, 0, 0, 0, 0, 0, 0, 0};
    v8i acc[4][4];
    #pragma unroll
    for (int mi = 0; mi < 4; ++mi)
        #pragma unroll
        for (int ni = 0; ni < 4; ++ni) acc[mi][ni] = vzero;

    // ---- tile copy: 6 x async b128 per thread per K-step (A 8KB + B 16KB) ----
    auto issue_copies = [&](char* Adst, char* Bdst, int kk) {
        #pragma unroll
        for (int j = 0; j < 2; ++j) {          // A: 128x64 = 512 chunks
            int c = tid + j * 256;
            int r = c >> 2, o = (c & 3) * 16;
            async_copy16(q8 + (size_t)(mBase + r) * K_DIM + kk + o, Adst + r * 64 + o);
        }
        #pragma unroll
        for (int j = 0; j < 4; ++j) {          // B: 256x64 = 1024 chunks
            int c = tid + j * 256;
            int r = c >> 2, o = (c & 3) * 16;
            async_copy16(w8 + (size_t)(nBase + r) * K_DIM + kk + o, Bdst + r * 64 + o);
        }
    };

    // 2-deep prefetch prologue
    issue_copies(As[0], Bs[0], 0);
    issue_copies(As[1], Bs[1], 64);
    wait_async<6>();                 // tile 0 resident, tile 1 may be in flight
    __syncthreads();

    int cur = 0, pre = 2;            // pre = buffer receiving tile it+2
    for (int it = 0; it < NIT; ++it) {
        if (it + 2 < NIT)
            issue_copies(As[pre], Bs[pre], (it + 2) * 64);

        // A fragments: 16x64 int8; lanes 0-15 hold K-chunks {0,16,32,48},
        // lanes 16-31 the +8 offsets (ISA 7.12.2 8-bit A layout)
        v8i afr[4];
        #pragma unroll
        for (int mi = 0; mi < 4; ++mi) {
            const char* p = As[cur] + (waveM * 64 + mi * 16 + l16) * 64 + half * 8;
            int2 t0 = *(const int2*)(p);
            int2 t1 = *(const int2*)(p + 16);
            int2 t2 = *(const int2*)(p + 32);
            int2 t3 = *(const int2*)(p + 48);
            afr[mi] = (v8i){t0.x, t0.y, t1.x, t1.y, t2.x, t2.y, t3.x, t3.y};
        }
        #pragma unroll
        for (int ni = 0; ni < 4; ++ni) {
            const char* p = Bs[cur] + (waveN * 64 + ni * 16 + l16) * 64 + half * 8;
            int2 t0 = *(const int2*)(p);
            int2 t1 = *(const int2*)(p + 16);
            int2 t2 = *(const int2*)(p + 32);
            int2 t3 = *(const int2*)(p + 48);
            v8i bfr = (v8i){t0.x, t0.y, t1.x, t1.y, t2.x, t2.y, t3.x, t3.y};
            #pragma unroll
            for (int mi = 0; mi < 4; ++mi)
                acc[mi][ni] = __builtin_amdgcn_wmma_i32_16x16x64_iu8(
                    true, afr[mi], true, bfr, acc[mi][ni], false, false);
        }

        // partial drain: next tile must be resident, newest may stay in flight
        if (it + 2 < NIT) wait_async<6>();
        else              wait_async<0>();
        __syncthreads();
        cur = (cur == 2) ? 0 : cur + 1;
        pre = (pre == 2) ? 0 : pre + 1;
    }

    // epilogue: y = (acc - zp*rowsum[n]) * scale * wscale[n] + bias[n]
    const float scale = scalezp[0];
    const int zpi = (int)scalezp[1];
    #pragma unroll
    for (int ni = 0; ni < 4; ++ni) {
        const int n = nBase + waveN * 64 + ni * 16 + l16;
        const float fs = scale * wscale[n];
        const float bv = bias[n];
        const int corr = zpi * rowsum[n];
        #pragma unroll
        for (int mi = 0; mi < 4; ++mi) {
            const int mb = mBase + waveM * 64 + mi * 16 + half * 8;
            #pragma unroll
            for (int r = 0; r < 8; ++r) {
                out[(size_t)(mb + r) * N_DIM + n] = (float)(acc[mi][ni][r] - corr) * fs + bv;
            }
        }
    }
}

extern "C" void kernel_launch(void* const* d_in, const int* in_sizes, int n_in,
                              void* d_out, int out_size, void* d_ws, size_t ws_size,
                              hipStream_t stream) {
    const float* x      = (const float*)d_in[0];   // [4,2048,4096]
    const float* weight = (const float*)d_in[1];   // [4096,4096] int8-valued f32
    const float* wscale = (const float*)d_in[2];   // [1,1,4096]
    const float* bias   = (const float*)d_in[3];   // [4096]
    float* out = (float*)d_out;

    char* ws = (char*)d_ws;
    char* q8 = ws;                                                  // 32 MB int8
    char* w8 = ws + (size_t)M_DIM * K_DIM;                          // 16 MB int8
    int* rowsum = (int*)(ws + (size_t)M_DIM * K_DIM + (size_t)N_DIM * K_DIM);
    float* scalezp  = (float*)((char*)rowsum + N_DIM * sizeof(int));
    float* partials = scalezp + 16;                                 // 2*NPART floats

    k_minmax<<<NPART, 256, 0, stream>>>(x, (M_DIM * K_DIM) / 4, partials);
    k_finalize<<<1, 256, 0, stream>>>(partials, NPART, scalezp);
    k_quant_x<<<4096, 256, 0, stream>>>(x, scalezp, (M_DIM * K_DIM) / 4, (int*)q8);
    k_quant_w<<<N_DIM, 256, 0, stream>>>(weight, (int*)w8, rowsum);
    k_gemm<<<dim3(N_DIM / 256, M_DIM / 128), 256, 0, stream>>>(
        q8, w8, rowsum, scalezp, wscale, bias, out);
}
